// STRankLossPair_24429773979784
// MI455X (gfx1250) — compile-verified, matrix operand order checked by reading, not run on hardware
//
#include <hip/hip_runtime.h>
#include <hip/hip_bf16.h>

// ---------------------------------------------------------------------------
// STRankLoss on MI455X (gfx1250, wave32)
//   Stage 2-4 build the segment-local permutation on device (hist/scan/scatter).
//   Stage 5 is the bandwidth-critical gather+log-softmax+reduce:
//     - per-wave double-buffered ASYNC gather (global_load_async_to_lds_b128,
//       throttled with s_wait_asynccnt),
//     - per-block partial reduction on the matrix pipe:
//       v_wmma_f32_16x16x4_f32 (exact f32) if declared, else the confirmed
//       v_wmma_f32_16x16x32_f16 ones-matrix trick.
// Roofline: ~512MB gathered reads -> ~22us at 23.3 TB/s; everything else is
// noise, so the design goal is purely latency-hiding on the random gather.
// ---------------------------------------------------------------------------

#define NGROUP 8192
#define FDIM   64
#define KCOPY  2
#define WAVE   32

typedef __attribute__((ext_vector_type(16))) _Float16 v16h;
typedef __attribute__((ext_vector_type(8)))  float    v8f;
typedef __attribute__((ext_vector_type(2)))  float    v2f;

// ---- feature probes -------------------------------------------------------
#if defined(__has_builtin)
#if __has_builtin(__builtin_amdgcn_global_load_async_to_lds_b128)
#define HAVE_ASYNC_LDS 1
#endif
#if __has_builtin(__builtin_amdgcn_wmma_f32_16x16x4_f32)
#define HAVE_WMMA_F32X4 1
#endif
#if __has_builtin(__builtin_amdgcn_s_wait_asynccnt)
#define HAVE_WAIT_ASYNC_BUILTIN 1
#endif
#endif

// Exact parameter types from the clang diagnostic:
//   param0: int __attribute__((vector_size(16))) __device__ *   (AS1 int4*)
//   param1: the AS3 counterpart
typedef int v4i __attribute__((vector_size(16)));
typedef __attribute__((address_space(1))) v4i gbl_v4i;
typedef __attribute__((address_space(3))) v4i lds_v4i;

__device__ __forceinline__ gbl_v4i* to_gbl(const void* p) {
    return (gbl_v4i*)(unsigned long long)p;
}
__device__ __forceinline__ lds_v4i* to_lds(void* p) {
    return (lds_v4i*)(unsigned)(unsigned long long)p;   // AS3 ptr = low 32 bits
}

__device__ __forceinline__ void wait_async_le2() {
#if defined(HAVE_WAIT_ASYNC_BUILTIN)
    __builtin_amdgcn_s_wait_asynccnt(2);
#else
    asm volatile("s_wait_asynccnt 0x2" ::: "memory");
#endif
}
__device__ __forceinline__ void wait_async_le0() {
#if defined(HAVE_WAIT_ASYNC_BUILTIN)
    __builtin_amdgcn_s_wait_asynccnt(0);
#else
    asm volatile("s_wait_asynccnt 0x0" ::: "memory");
#endif
}

__device__ __forceinline__ unsigned mix32(unsigned x) {
    x ^= x >> 16; x *= 0x7feb352du;
    x ^= x >> 15; x *= 0x846ca68bu;
    x ^= x >> 16; return x;
}

__device__ __forceinline__ float pair_term(float a, float b, float ca, float cb) {
    float m   = fmaxf(a, b);
    float d   = fminf(a, b) - m;                 // <= 0, exp can't overflow
    float lse = m + __logf(1.0f + __expf(d));
    return (lse - a) * ca + (lse - b) * cb;
}

// Resolve the two permuted source rows for sorted position p.
__device__ __forceinline__ void compute_srcs(
        int p, const int* __restrict__ sorted_idx, const int* __restrict__ pos_group,
        const int* __restrict__ offsets, const int* __restrict__ hist,
        int& src0, int& src1) {
    int g    = pos_group[p];
    int base = offsets[g];
    int size = hist[g];                           // >= 1 for any live position
    int i    = p - base;
    int r0 = (int)(mix32((unsigned)g * 0x9E3779B9u + 0x2545F491u) % (unsigned)size);
    int r1 = (int)(mix32((unsigned)g * 0x9E3779B9u + 0x85EBCA6Bu) % (unsigned)size);
    int j0 = i + r0; if (j0 >= size) j0 -= size;
    int j1 = i + r1; if (j1 >= size) j1 -= size;
    src0 = sorted_idx[base + j0];
    src1 = sorted_idx[base + j1];
}

// ---------------- Stage 1: zero ----------------
__global__ void zero_u32_kernel(unsigned* __restrict__ p, int n) {
    int i = blockIdx.x * blockDim.x + threadIdx.x;
    if (i < n) p[i] = 0u;
}

// ---------------- Stage 2: histogram ----------------
__global__ void hist_kernel(const int* __restrict__ groups, int* __restrict__ hist, int n) {
    for (int i = blockIdx.x * blockDim.x + threadIdx.x; i < n; i += gridDim.x * blockDim.x)
        atomicAdd(&hist[groups[i]], 1);
}

// ---------------- Stage 3: exclusive scan of NGROUP bins ----------------
__global__ __launch_bounds__(1024) void scan_kernel(const int* __restrict__ hist,
                                                    int* __restrict__ offsets) {
    __shared__ int sums[1024];
    const int t = threadIdx.x;                    // 1024 threads, 8 bins each
    int v[8]; int s = 0;
#pragma unroll
    for (int j = 0; j < 8; ++j) { v[j] = hist[t * 8 + j]; s += v[j]; }
    sums[t] = s;
    __syncthreads();
    for (int off = 1; off < 1024; off <<= 1) {
        int x = (t >= off) ? sums[t - off] : 0;
        __syncthreads();
        sums[t] += x;
        __syncthreads();
    }
    int run = (t > 0) ? sums[t - 1] : 0;
#pragma unroll
    for (int j = 0; j < 8; ++j) { offsets[t * 8 + j] = run; run += v[j]; }
}

// ---------------- Stage 4: scatter into group-contiguous order ----------------
__global__ void scatter_kernel(const int* __restrict__ groups,
                               const int* __restrict__ offsets,
                               int* __restrict__ cursor,
                               int* __restrict__ sorted_idx,
                               int* __restrict__ pos_group, int n) {
    for (int i = blockIdx.x * blockDim.x + threadIdx.x; i < n; i += gridDim.x * blockDim.x) {
        int g = groups[i];
        int r = atomicAdd(&cursor[g], 1);
        int p = offsets[g] + r;
        sorted_idx[p] = i;
        pos_group[p]  = g;
    }
}

// Per-block reduction on the matrix pipe + single f32 atomic.
__device__ __forceinline__ void block_reduce_wmma(float acc, float* partials,
                                                  float* accum) {
    partials[threadIdx.x] = acc;
    __syncthreads();
    if (threadIdx.x < WAVE) {                     // wave 0, EXEC all-ones
        const int lane = threadIdx.x;
        float v;
#if defined(HAVE_WMMA_F32X4)
        // Exact f32: D += ones(16x4) x B(4x16); 4 chunks fold 256 partials.
        v2f a; a[0] = 1.0f; a[1] = 1.0f;
        v8f c = {};
#pragma unroll
        for (int t = 0; t < 4; ++t) {
            v2f b;
            b[0] = partials[t * 64 + lane];
            b[1] = partials[t * 64 + 32 + lane];
            c = __builtin_amdgcn_wmma_f32_16x16x4_f32(
                false, a, false, b, (short)0, c, false, false);
        }
        v = c[0];
#else
        // Confirmed fallback: ones(16x32,f16) x B collapses all of B.
        v16h ones;
#pragma unroll
        for (int h = 0; h < 16; ++h) ones[h] = (_Float16)1.0f;
        v16h bmat;
#pragma unroll
        for (int h = 0; h < 16; ++h)
            bmat[h] = (h < 8) ? (_Float16)partials[lane * 8 + h] : (_Float16)0.0f;
        v8f c = {};
        c = __builtin_amdgcn_wmma_f32_16x16x32_f16(
            false, ones, false, bmat, (short)0, c, false, false);
        v = c[0];
#endif
#pragma unroll
        for (int m = 16; m >= 1; m >>= 1) v += __shfl_xor(v, m, WAVE);
        if (lane == 0) atomicAdd(accum, v * 0.5f);  // rows duplicated -> *0.5
    }
}

// ---------------- Stage 5: fused gather + log-softmax + reduce ----------------
#if defined(HAVE_ASYNC_LDS)

// Issue one stage: 2 async b128 ops move 4 rows (1KB) into this wave's buffer.
// Lane layout: lane<16 -> row src0 chunk `lane`; lane>=16 -> row src1.
__device__ __forceinline__ void issue_stage(const float* __restrict__ pred,
                                            const float* __restrict__ count,
                                            int src0, int src1,
                                            char* lbuf, int lane) {
    const int   sel   = lane >> 4;
    const int   chunk = lane & 15;
    const size_t row  = (size_t)(sel ? src1 : src0) * FDIM;
    __builtin_amdgcn_global_load_async_to_lds_b128(
        to_gbl(pred + row + chunk * 4),  to_lds(lbuf + lane * 16),       0, 0);
    __builtin_amdgcn_global_load_async_to_lds_b128(
        to_gbl(count + row + chunk * 4), to_lds(lbuf + 512 + lane * 16), 0, 0);
}

__global__ __launch_bounds__(256) void loss_kernel(
        const float* __restrict__ pred,  const float* __restrict__ count,
        const int* __restrict__ sorted_idx, const int* __restrict__ pos_group,
        const int* __restrict__ offsets,    const int* __restrict__ hist,
        float* __restrict__ accum, int n) {
    __shared__ __align__(16) char abuf[8 * 2 * 1024];   // 8 waves x double 1KB
    __shared__ float partials[256];
    const int lane    = threadIdx.x & (WAVE - 1);
    const int wave    = threadIdx.x >> 5;
    const int wstride = gridDim.x * 8;
    char* wbuf = abuf + wave * 2048;

    float acc = 0.0f;
    int   p   = blockIdx.x * 8 + wave;
    int   par = 0;
    if (p < n) {
        int s0, s1;
        compute_srcs(p, sorted_idx, pos_group, offsets, hist, s0, s1);
        issue_stage(pred, count, s0, s1, wbuf, lane);
    }
    for (; p < n; p += wstride) {
        int pn = p + wstride;
        if (pn < n) {
            if (pn + wstride < n)                       // global_prefetch_b8
                __builtin_prefetch(&pos_group[pn + wstride], 0, 1);
            int t0, t1;
            compute_srcs(pn, sorted_idx, pos_group, offsets, hist, t0, t1);
            issue_stage(pred, count, t0, t1, wbuf + (par ^ 1) * 1024, lane);
            wait_async_le2();                           // stage p landed
        } else {
            wait_async_le0();                           // drain
        }
        asm volatile("" ::: "memory");                  // no LDS-read hoisting

        const float2* bp = (const float2*)(wbuf + par * 1024);
        float2 a  = bp[lane];                           // pred[src0]
        float2 b  = bp[32 + lane];                      // pred[src1]
        float2 wa = bp[64 + lane];                      // count[src0]
        float2 wb = bp[96 + lane];                      // count[src1]
        acc += pair_term(a.x, b.x, wa.x, wb.x);
        acc += pair_term(a.y, b.y, wa.y, wb.y);
        par ^= 1;
    }
    block_reduce_wmma(acc, partials, accum);
}

#else  // synchronous fallback

__global__ __launch_bounds__(256) void loss_kernel(
        const float* __restrict__ pred,  const float* __restrict__ count,
        const int* __restrict__ sorted_idx, const int* __restrict__ pos_group,
        const int* __restrict__ offsets,    const int* __restrict__ hist,
        float* __restrict__ accum, int n) {
    __shared__ float partials[256];
    const int lane    = threadIdx.x & (WAVE - 1);
    const int wave    = threadIdx.x >> 5;
    const int wstride = gridDim.x * 8;

    float acc = 0.0f;
    for (int p = blockIdx.x * 8 + wave; p < n; p += wstride) {
        if (p + wstride < n)
            __builtin_prefetch(&pos_group[p + wstride], 0, 1);
        int src0, src1;
        compute_srcs(p, sorted_idx, pos_group, offsets, hist, src0, src1);
        const float2* pa = (const float2*)(pred  + (size_t)src0 * FDIM);
        const float2* pb = (const float2*)(pred  + (size_t)src1 * FDIM);
        const float2* ca = (const float2*)(count + (size_t)src0 * FDIM);
        const float2* cb = (const float2*)(count + (size_t)src1 * FDIM);
        float2 a  = pa[lane];
        float2 b  = pb[lane];
        float2 wa = ca[lane];
        float2 wb = cb[lane];
        acc += pair_term(a.x, b.x, wa.x, wb.x);
        acc += pair_term(a.y, b.y, wa.y, wb.y);
    }
    block_reduce_wmma(acc, partials, accum);
}

#endif

// ---------------- Stage 6: finalize ----------------
__global__ void finalize_kernel(const float* __restrict__ accum,
                                float* __restrict__ out, float inv_n) {
    if (blockIdx.x == 0 && threadIdx.x == 0) out[0] = accum[0] * inv_n;
}

extern "C" void kernel_launch(void* const* d_in, const int* in_sizes, int n_in,
                              void* d_out, int out_size, void* d_ws, size_t ws_size,
                              hipStream_t stream) {
    const float* pred   = (const float*)d_in[0];
    const float* count  = (const float*)d_in[1];
    const int*   groups = (const int*)d_in[2];
    const int N = in_sizes[2];                       // 500000
    float* out = (float*)d_out;

    int* hist       = (int*)d_ws;                    // NGROUP
    int* cursor     = hist + NGROUP;                 // NGROUP (adjacent: one zero)
    int* offsets    = cursor + NGROUP;               // NGROUP
    int* pos_group  = offsets + NGROUP;              // N
    int* sorted_idx = pos_group + N;                 // N
    float* accum    = (float*)(sorted_idx + N);      // 1

    zero_u32_kernel<<<(2 * NGROUP + 255) / 256, 256, 0, stream>>>((unsigned*)hist, 2 * NGROUP);
    zero_u32_kernel<<<1, 32, 0, stream>>>((unsigned*)accum, 1);

    hist_kernel<<<512, 256, 0, stream>>>(groups, hist, N);
    scan_kernel<<<1, 1024, 0, stream>>>(hist, offsets);
    scatter_kernel<<<512, 256, 0, stream>>>(groups, offsets, cursor, sorted_idx, pos_group, N);

    loss_kernel<<<4096, 256, 0, stream>>>(pred, count, sorted_idx, pos_group,
                                          offsets, hist, accum, N);

    const float inv_n = 1.0f / ((float)N * (float)KCOPY * (float)FDIM);
    finalize_kernel<<<1, 32, 0, stream>>>(accum, out, inv_n);
}
